// MultiHeadAttention_81063212745174
// MI455X (gfx1250) — compile-verified, matrix operand order checked by reading, not run on hardware
//
#include <hip/hip_runtime.h>

// ---------------------------------------------------------------------------
// MI455X (gfx1250) multi-head attention:
//   bf16 WMMA GEMMs (TDM tensor_load_to_lds staging, TENSORcnt double-buffer)
//   + flash attention (async global->LDS staging, ASYNCcnt double-buffer).
// ---------------------------------------------------------------------------

typedef __bf16 bf16_t;
typedef __attribute__((ext_vector_type(16))) __bf16 v16bf;
typedef __attribute__((ext_vector_type(8)))  __bf16 v8bf;
typedef __attribute__((ext_vector_type(4)))  __bf16 v4bf;
typedef __attribute__((ext_vector_type(8)))  float  v8f;
typedef __attribute__((ext_vector_type(4)))  float  v4f32;
typedef __attribute__((ext_vector_type(4)))  unsigned int u32x4;
typedef __attribute__((ext_vector_type(8)))  int i32x8;
typedef __attribute__((ext_vector_type(4)))  int i32x4;

constexpr int NB  = 4;     // batch
constexpr int SQ  = 2048;  // query length
constexpr int TK  = 2048;  // key length
constexpr int EMB = 1024;  // embedding
constexpr int NH  = 16;    // heads
constexpr int HD  = 64;    // head dim
constexpr int MROW = NB * SQ;  // 8192 rows for the projection GEMMs

#if __has_builtin(__builtin_amdgcn_tensor_load_to_lds)
#define HAVE_TDM 1
#else
#define HAVE_TDM 0
#endif

// ----------------------------- helpers -------------------------------------

__device__ __forceinline__ bf16_t to_bf16(float f) {
  unsigned u = __builtin_bit_cast(unsigned, f);
  u += 0x7fffu + ((u >> 16) & 1u);            // round to nearest even
  unsigned short h = (unsigned short)(u >> 16);
  return __builtin_bit_cast(bf16_t, h);
}

__device__ __forceinline__ v8f wmma_bf16(v16bf a, v16bf b, v8f c) {
  // (neg_a, A, neg_b, B, c_mod, C, reuse_a, reuse_b)
  return __builtin_amdgcn_wmma_f32_16x16x32_bf16(false, a, false, b, (short)0, c,
                                                 false, false);
}

// Low 32 bits of a generic LDS pointer == LDS byte address (ISA 10.2).
__device__ __forceinline__ unsigned lds_addr(const void* p) {
  return (unsigned)(uintptr_t)p;
}

// CDNA5 async copy: 16B per lane, global -> LDS, tracked by ASYNCcnt.
__device__ __forceinline__ void async_copy_b128(unsigned lds_off, const void* gsrc) {
  asm volatile("global_load_async_to_lds_b128 %0, %1, off"
               :
               : "v"(lds_off), "v"(gsrc)
               : "memory");
}
#define WAIT_ASYNC(n) asm volatile("s_wait_asynccnt " #n ::: "memory")

#if HAVE_TDM
// TDM descriptor load: 128 rows x 32 cols of bf16, row-major source with
// stride 1024 elements, LDS destination padded by the TDM itself to a 40-
// element row stride (pad_interval = 16 DWORDs, pad_amount = 4 DWORDs).
// amdgpu-toolchain (clang-23) 6-arg builtin form:
//   (uint32x4 g0, int32x8 g1, int32x4 g2, int32x4 g3, int32x8 g4, i32 cpol)
__device__ __forceinline__ void tdm_load_tile_128x32(const bf16_t* gsrc,
                                                     unsigned lds_byte) {
  const unsigned long long ga = (unsigned long long)(uintptr_t)gsrc;
  u32x4 g0 = {};
  g0[0] = 1u;                                                 // count=1, user D#
  g0[1] = lds_byte;                                           // lds_addr
  g0[2] = (unsigned)(ga & 0xffffffffu);                       // global_addr lo
  g0[3] = (unsigned)((ga >> 32) & 0x01ffffffu) | (2u << 30);  // addr[56:32]|type=2
  i32x8 g1 = {};
  g1[0] = (1 << 16)        // data_size = 2 bytes
        | (1 << 20)        // pad_enable
        | (3 << 22)        // pad_interval: 16 DWORDs (one 32-elem bf16 row)
        | (3 << 25);       // pad_amount: 4 DWORDs (8 bf16) -> LDS stride 40
  g1[1] = (int)((1024u & 0xffffu) << 16);  // tensor_dim0 = 1024 (low 16)
  g1[2] = (int)(0xffffu << 16);            // dim0 hi = 0, tensor_dim1 = 65535
  g1[3] = (32 << 16);                      // dim1 hi = 0, tile_dim0 = 32
  g1[4] = 128;                             // tile_dim1 = 128, tile_dim2 = 0
  g1[5] = 1024;                            // tensor_dim0_stride = 1024 (lo 32)
  g1[6] = 0;
  g1[7] = 0;
  i32x4 g2 = {};
  i32x4 g3 = {};
  i32x8 g4 = {};
  __builtin_amdgcn_tensor_load_to_lds(g0, g1, g2, g3, g4, 0);
}
#endif

// A-layout fragment load (16x32 bf16, K=32) from a row-major [rows][rowStride]
// surface. lane L<16 holds row L, K chunks {0..7,16..23}; lane>=16 holds
// K chunks {8..15,24..31} (per CDNA5 ISA 7.12.2).
__device__ __forceinline__ v16bf load_frag(const bf16_t* base, int lane,
                                           int rowStride, int kOff) {
  const bf16_t* p = base + (size_t)(lane & 15) * rowStride + kOff + ((lane >> 4) << 3);
  v8bf lo = *(const v8bf*)(p);
  v8bf hi = *(const v8bf*)(p + 16);
  return __builtin_shufflevector(lo, hi, 0, 1, 2, 3, 4, 5, 6, 7,
                                 8, 9, 10, 11, 12, 13, 14, 15);
}

// XOR-swizzle reductions within each 16-lane half (wave32 C-layout: a row's
// 16 columns live across lanes 0-15 or 16-31 of one VGPR).
template <int MASK>
__device__ __forceinline__ float swz_xor(float x) {
  int i = __builtin_bit_cast(int, x);
  int j = __builtin_amdgcn_ds_swizzle(i, (MASK << 10) | 0x1f);
  return __builtin_bit_cast(float, j);
}
__device__ __forceinline__ float rmax16(float x) {
  x = fmaxf(x, swz_xor<1>(x));
  x = fmaxf(x, swz_xor<2>(x));
  x = fmaxf(x, swz_xor<4>(x));
  x = fmaxf(x, swz_xor<8>(x));
  return x;
}
__device__ __forceinline__ float rsum16(float x) {
  x += swz_xor<1>(x);
  x += swz_xor<2>(x);
  x += swz_xor<4>(x);
  x += swz_xor<8>(x);
  return x;
}

// ------------------------- f32 -> bf16 convert (x4) -------------------------

__global__ __launch_bounds__(256) void cvt_f32_bf16(const float* __restrict__ s,
                                                    bf16_t* __restrict__ d, int n4) {
  int i = blockIdx.x * 256 + threadIdx.x;
  if (i < n4) {
    v4f32 v = *(const v4f32*)(s + (size_t)i * 4);
    v4bf o;
    o[0] = to_bf16(v[0]); o[1] = to_bf16(v[1]);
    o[2] = to_bf16(v[2]); o[3] = to_bf16(v[3]);
    *(v4bf*)(d + (size_t)i * 4) = o;
  }
}

// ------------------------------ GEMM ----------------------------------------
// C[m,o] = sum_k A[m,k] * B[o,k] + bias[o]   (torch Linear: x @ W^T + b)
// Block tile 128x128, 8 waves (4x2), each wave 32x64 (2x4 WMMA tiles), K-step
// 32, double-buffered LDS fed by the Tensor Data Mover (2 descriptors/stage
// issued by wave 0, TENSORcnt pipelined) or async-to-LDS fallback.
// MODE 0: out bf16, head-major [N,H,S,64]     (Q, K projections)
// MODE 1: out bf16, transposed [N,H,64,T]     (V projection)
// MODE 2: out f32, plain [M,E]                (final projection -> d_out)
template <int MODE>
__global__ __launch_bounds__(256) void gemm_proj(const bf16_t* __restrict__ A,
                                                 const bf16_t* __restrict__ B,
                                                 const float* __restrict__ bias,
                                                 void* __restrict__ outp) {
  constexpr int K = EMB;
  constexpr int KT = K / 32;   // 32 k-steps
  constexpr int LDK = 40;      // padded LDS row (bf16 elements)
  __shared__ bf16_t lA[2][128 * LDK];
  __shared__ bf16_t lB[2][128 * LDK];

  const int tid  = threadIdx.x;
  const int lane = tid & 31;
  const int wave = tid >> 5;
  const int wm = wave >> 1;  // 0..3 -> 32-row strip
  const int wn = wave & 1;   // 0..1 -> 64-col strip
  const int tileM = blockIdx.y * 128;
  const int tileN = blockIdx.x * 128;

#if HAVE_TDM
  const bf16_t* gAtile = A + (size_t)tileM * K;
  const bf16_t* gBtile = B + (size_t)tileN * K;
  auto issue_stage = [&](int kt, int buf) {
    if (wave == 0) {
      tdm_load_tile_128x32(gAtile + kt * 32, lds_addr(&lA[buf][0]));
      tdm_load_tile_128x32(gBtile + kt * 32, lds_addr(&lB[buf][0]));
    }
  };
#else
  // Each thread moves two 16B chunks per matrix per stage.
  const int crow = tid >> 2;       // 0..63
  const int csub = tid & 3;        // 16B sub-chunk within a 64B row
  const bf16_t* gA0 = A + (size_t)(tileM + crow) * K + csub * 8;
  const bf16_t* gA1 = A + (size_t)(tileM + 64 + crow) * K + csub * 8;
  const bf16_t* gB0 = B + (size_t)(tileN + crow) * K + csub * 8;
  const bf16_t* gB1 = B + (size_t)(tileN + 64 + crow) * K + csub * 8;
  const int sOff0 = crow * LDK + csub * 8;
  const int sOff1 = (64 + crow) * LDK + csub * 8;
  auto issue_stage = [&](int kt, int buf) {
    const int ko = kt * 32;
    async_copy_b128(lds_addr(&lA[buf][sOff0]), gA0 + ko);
    async_copy_b128(lds_addr(&lA[buf][sOff1]), gA1 + ko);
    async_copy_b128(lds_addr(&lB[buf][sOff0]), gB0 + ko);
    async_copy_b128(lds_addr(&lB[buf][sOff1]), gB1 + ko);
  };
#endif

  issue_stage(0, 0);

  v8f acc[2][4] = {};

  for (int kt = 0; kt < KT; ++kt) {
    const int cur = kt & 1;
    if (kt + 1 < KT) {
      issue_stage(kt + 1, cur ^ 1);   // keep next stage in flight
#if HAVE_TDM
      __builtin_amdgcn_s_wait_tensorcnt(2);
#else
      WAIT_ASYNC(4);
#endif
    } else {
#if HAVE_TDM
      __builtin_amdgcn_s_wait_tensorcnt(0);
#else
      WAIT_ASYNC(0);
#endif
    }
    __syncthreads();

    const bf16_t* la = &lA[cur][0];
    const bf16_t* lb = &lB[cur][0];
    v16bf af[2], bfr[4];
#pragma unroll
    for (int i = 0; i < 2; ++i)
      af[i] = load_frag(la + (wm * 32 + i * 16) * LDK, lane, LDK, 0);
#pragma unroll
    for (int j = 0; j < 4; ++j)
      bfr[j] = load_frag(lb + (wn * 64 + j * 16) * LDK, lane, LDK, 0);
#pragma unroll
    for (int i = 0; i < 2; ++i)
#pragma unroll
      for (int j = 0; j < 4; ++j)
        acc[i][j] = wmma_bf16(af[i], bfr[j], acc[i][j]);
    __syncthreads();  // everyone done reading `cur` before it is refilled
  }

  // Epilogue: C layout -> VGPR g holds row (g + 8*(lane/16)), col = lane%16.
#pragma unroll
  for (int i = 0; i < 2; ++i) {
    const int mrow0 = tileM + wm * 32 + i * 16 + ((lane >> 4) << 3);
#pragma unroll
    for (int j = 0; j < 4; ++j) {
      const int ocol = tileN + wn * 64 + j * 16 + (lane & 15);
      const float bv = bias[ocol];
#pragma unroll
      for (int g = 0; g < 8; ++g) {
        const float v = acc[i][j][g] + bv;
        const int m = mrow0 + g;
        if constexpr (MODE == 2) {
          ((float*)outp)[(size_t)m * EMB + ocol] = v;
        } else {
          const int bn = m >> 11, s = m & (SQ - 1);
          const int hh = ocol >> 6, d = ocol & (HD - 1);
          bf16_t* o = (bf16_t*)outp;
          if constexpr (MODE == 0)
            o[(((size_t)(bn * NH + hh)) * SQ + s) * HD + d] = to_bf16(v);
          else
            o[(((size_t)(bn * NH + hh)) * HD + d) * TK + s] = to_bf16(v);
        }
      }
    }
  }
}

// --------------------------- flash attention --------------------------------
// grid = (S/64, H, N), 128 threads (4 waves). Each wave owns 16 query rows,
// streams keys in 64-wide chunks with online softmax; the 64x64 K and V tiles
// are async-staged into LDS once per block (double buffered, ASYNCcnt) and
// shared by all 4 waves. Scores never touch HBM.
__global__ __launch_bounds__(128) void attn_fwd(const bf16_t* __restrict__ Q,
                                                const bf16_t* __restrict__ Kp,
                                                const bf16_t* __restrict__ Vt,
                                                bf16_t* __restrict__ Y) {
  constexpr int LDT = 72;                 // padded LDS row (bf16 elements)
  __shared__ bf16_t lK[2][64 * LDT];      // keys:  row = key idx, col = d
  __shared__ bf16_t lV[2][64 * LDT];      // V^T:   row = d,       col = key idx
  __shared__ bf16_t lP[4][16][LDT];       // per-wave P transpose staging

  const int tid  = threadIdx.x;
  const int lane = tid & 31;
  const int wave = tid >> 5;
  const int qt = blockIdx.x;
  const int h  = blockIdx.y;
  const int n  = blockIdx.z;

  const size_t headQ = ((size_t)(n * NH + h)) * SQ * HD;
  const size_t headK = ((size_t)(n * NH + h)) * TK * HD;
  const size_t headV = ((size_t)(n * NH + h)) * HD * TK;
  const int q0 = qt * 64 + wave * 16;

  const v16bf qf0 = load_frag(Q + headQ + (size_t)q0 * HD, lane, HD, 0);
  const v16bf qf1 = load_frag(Q + headQ + (size_t)q0 * HD, lane, HD, 32);

  // Async staging: 64x64 bf16 tile = 512 x 16B chunks; 128 threads x 4 chunks.
  auto issue_stage = [&](int kb0, int buf) {
#pragma unroll
    for (int i = 0; i < 4; ++i) {
      const int c = tid + 128 * i;       // 0..511
      const int row = c >> 3;            // 0..63
      const int sub = c & 7;             // 16B chunk within 128B row
      async_copy_b128(lds_addr(&lK[buf][row * LDT + sub * 8]),
                      Kp + headK + (size_t)(kb0 + row) * HD + sub * 8);
      async_copy_b128(lds_addr(&lV[buf][row * LDT + sub * 8]),
                      Vt + headV + (size_t)row * TK + kb0 + sub * 8);
    }
  };

  issue_stage(0, 0);  // ASYNCcnt = 8

  v8f acc[4] = {};
  float mrun[8], lrun[8];
#pragma unroll
  for (int g = 0; g < 8; ++g) { mrun[g] = -1e30f; lrun[g] = 0.f; }

  for (int kb0 = 0; kb0 < TK; kb0 += 64) {
    const int cur = (kb0 >> 6) & 1;
    if (kb0 + 64 < TK) {
      issue_stage(kb0 + 64, cur ^ 1);  // ASYNCcnt up to 16
      WAIT_ASYNC(8);
    } else {
      WAIT_ASYNC(0);
    }
    __syncthreads();

    // S = (Q K^T) * scale : four 16x16 tiles, contraction 64 = 2 WMMA steps.
    v8f sc[4];
#pragma unroll
    for (int j = 0; j < 4; ++j) {
      const bf16_t* kbase = &lK[cur][(j * 16) * LDT];
      const v16bf kf0 = load_frag(kbase, lane, LDT, 0);
      const v16bf kf1 = load_frag(kbase, lane, LDT, 32);
      v8f s = {};
      s = wmma_bf16(qf0, kf0, s);
      s = wmma_bf16(qf1, kf1, s);
#pragma unroll
      for (int g = 0; g < 8; ++g) sc[j][g] = s[g] * 0.125f;  // 1/sqrt(64)
    }

    // Online softmax bookkeeping (row = (g, lane/16); cols across 16 lanes).
    float mnew[8], corr[8], psum[8];
#pragma unroll
    for (int g = 0; g < 8; ++g) {
      float rm = fmaxf(fmaxf(sc[0][g], sc[1][g]), fmaxf(sc[2][g], sc[3][g]));
      rm = rmax16(rm);
      mnew[g] = fmaxf(mrun[g], rm);
      corr[g] = __expf(mrun[g] - mnew[g]);
      mrun[g] = mnew[g];
      psum[g] = 0.f;
    }

    // P = exp(S - m), store bf16 to LDS for C-layout -> A-layout transpose.
    const int prow = (lane >> 4) << 3;
    const int pcol = lane & 15;
#pragma unroll
    for (int j = 0; j < 4; ++j)
#pragma unroll
      for (int g = 0; g < 8; ++g) {
        const float p = __expf(sc[j][g] - mnew[g]);
        psum[g] += p;
        lP[wave][g + prow][j * 16 + pcol] = to_bf16(p);
      }
#pragma unroll
    for (int g = 0; g < 8; ++g) {
      lrun[g] = lrun[g] * corr[g] + rsum16(psum[g]);
#pragma unroll
      for (int jd = 0; jd < 4; ++jd) acc[jd][g] *= corr[g];
    }

    // Reload P as A operand (DS ops are in-order within a wave).
    const v16bf pf0 = load_frag(&lP[wave][0][0], lane, LDT, 0);
    const v16bf pf1 = load_frag(&lP[wave][0][0], lane, LDT, 32);

    // O += P @ V ; V^T tile in LDS: rows = d, cols = key idx.
#pragma unroll
    for (int jd = 0; jd < 4; ++jd) {
      const bf16_t* vbase = &lV[cur][(jd * 16) * LDT];
      const v16bf vf0 = load_frag(vbase, lane, LDT, 0);
      const v16bf vf1 = load_frag(vbase, lane, LDT, 32);
      acc[jd] = wmma_bf16(pf0, vf0, acc[jd]);
      acc[jd] = wmma_bf16(pf1, vf1, acc[jd]);
    }
    __syncthreads();  // all waves done with `cur` K/V before refill
  }

#pragma unroll
  for (int g = 0; g < 8; ++g) lrun[g] = 1.f / lrun[g];
  const int row0 = n * SQ + q0 + ((lane >> 4) << 3);
#pragma unroll
  for (int jd = 0; jd < 4; ++jd) {
    const int col = h * HD + jd * 16 + (lane & 15);
#pragma unroll
    for (int g = 0; g < 8; ++g)
      Y[(size_t)(row0 + g) * EMB + col] = to_bf16(acc[jd][g] * lrun[g]);
  }
}

// ------------------------------ launch --------------------------------------

extern "C" void kernel_launch(void* const* d_in, const int* in_sizes, int n_in,
                              void* d_out, int out_size, void* d_ws, size_t ws_size,
                              hipStream_t stream) {
  (void)in_sizes; (void)n_in; (void)out_size; (void)ws_size;
  const float* query = (const float*)d_in[0];
  const float* key   = (const float*)d_in[1];
  const float* value = (const float*)d_in[2];
  const float* Wq = (const float*)d_in[3];
  const float* bq = (const float*)d_in[4];
  const float* Wk = (const float*)d_in[5];
  const float* bk = (const float*)d_in[6];
  const float* Wv = (const float*)d_in[7];
  const float* bv = (const float*)d_in[8];
  const float* Wp = (const float*)d_in[9];
  const float* bp = (const float*)d_in[10];

  const size_t ACT = (size_t)MROW * EMB;  // 8.4M elements
  const size_t WEL = (size_t)EMB * EMB;   // 1M elements

  char* w = (char*)d_ws;
  auto take = [&](size_t elems) { void* p = w; w += elems * sizeof(bf16_t); return (bf16_t*)p; };
  bf16_t* qb  = take(ACT);
  bf16_t* kb  = take(ACT);
  bf16_t* vb  = take(ACT);
  bf16_t* wqb = take(WEL);
  bf16_t* wkb = take(WEL);
  bf16_t* wvb = take(WEL);
  bf16_t* wpb = take(WEL);
  bf16_t* Qp  = take(ACT);
  bf16_t* Kpj = take(ACT);
  bf16_t* Vt  = take(ACT);
  bf16_t* Yb  = qb;  // qb is dead after the Q projection; reuse for attn output

  const int na4 = (int)(ACT / 4), nw4 = (int)(WEL / 4);
  cvt_f32_bf16<<<(na4 + 255) / 256, 256, 0, stream>>>(query, qb, na4);
  cvt_f32_bf16<<<(na4 + 255) / 256, 256, 0, stream>>>(key, kb, na4);
  cvt_f32_bf16<<<(na4 + 255) / 256, 256, 0, stream>>>(value, vb, na4);
  cvt_f32_bf16<<<(nw4 + 255) / 256, 256, 0, stream>>>(Wq, wqb, nw4);
  cvt_f32_bf16<<<(nw4 + 255) / 256, 256, 0, stream>>>(Wk, wkb, nw4);
  cvt_f32_bf16<<<(nw4 + 255) / 256, 256, 0, stream>>>(Wv, wvb, nw4);
  cvt_f32_bf16<<<(nw4 + 255) / 256, 256, 0, stream>>>(Wp, wpb, nw4);

  dim3 gg(EMB / 128, MROW / 128);
  gemm_proj<0><<<gg, 256, 0, stream>>>(qb, wqb, bq, Qp);
  gemm_proj<0><<<gg, 256, 0, stream>>>(kb, wkb, bk, Kpj);
  gemm_proj<1><<<gg, 256, 0, stream>>>(vb, wvb, bv, Vt);

  dim3 ga(SQ / 64, NH, NB);
  attn_fwd<<<ga, 128, 0, stream>>>(Qp, Kpj, Vt, Yb);

  gemm_proj<2><<<gg, 256, 0, stream>>>(Yb, wpb, bp, d_out);
}
